// R_GNN_80109730005609
// MI455X (gfx1250) — matrix operand order repaired and checked by reading, not compiled, not run on hardware
//
#include <hip/hip_runtime.h>

// ---------------------------------------------------------------------------
// GCN link predictor for MI455X (gfx1250, wave32, WMMA).
//   h1 = relu(gcn(x, W1(128x16)) + b1)
//   h  = gcn(h1, W2(16x128)) + b2
//   out[p] = sigmoid(relu([h[s];h[t]] @ dw1(256x128) + db1) @ dw2(128) + db2)
// Decoder GEMM (65.5 GFLOP) runs on v_wmma_f32_16x16x32_bf16 with weight
// fragments pre-swizzled to the ISA lane layout (b128 loads, no per-element
// gather/convert in the hot kernels).
// ---------------------------------------------------------------------------

typedef __attribute__((ext_vector_type(16))) __bf16 v16bf;
typedef __attribute__((ext_vector_type(8)))  float  v8f;

struct alignas(16) U4 { unsigned int x, y, z, w; };

union BF16x16 {
    v16bf          v;
    unsigned short u[16];
    unsigned int   w[8];
    U4             q[2];
};

__device__ __forceinline__ unsigned short f2bf(float f) {
    unsigned int x = __float_as_uint(f);
    unsigned int r = x + 0x7FFFu + ((x >> 16) & 1u);   // round-to-nearest-even
    return (unsigned short)(r >> 16);
}

#if __has_builtin(__builtin_amdgcn_cvt_pk_bf16_f32)
typedef __attribute__((ext_vector_type(2))) __bf16 v2bf;
__device__ __forceinline__ unsigned int f2bf2(float a, float b) {  // a->lo, b->hi
    union { v2bf v; unsigned int u; } cv;
    cv.v = __builtin_amdgcn_cvt_pk_bf16_f32(a, b);
    return cv.u;
}
#else
__device__ __forceinline__ unsigned int f2bf2(float a, float b) {
    return (unsigned int)f2bf(a) | ((unsigned int)f2bf(b) << 16);
}
#endif

__device__ __forceinline__ v8f wmma_bf16(const BF16x16& a, const BF16x16& b, v8f c) {
    return __builtin_amdgcn_wmma_f32_16x16x32_bf16(
        /*neg_a=*/false, a.v, /*neg_b=*/false, b.v,
        /*c_mod=*/(short)0, c, /*reuse_a=*/false, /*reuse_b=*/false);
}

// ------------------------------ degree prep --------------------------------

__global__ void k_deg_init(float* deg, int n) {
    int i = blockIdx.x * blockDim.x + threadIdx.x;
    if (i < n) deg[i] = 1.0f;                       // self-loop contribution
}

__global__ void k_deg_count(const int* __restrict__ dst, float* deg, int e) {
    int i = blockIdx.x * blockDim.x + threadIdx.x;
    if (i < e) atomicAdd(&deg[dst[i]], 1.0f);
}

__global__ void k_deg_rsqrt(float* deg, int n) {   // deg -> dinv in place
    int i = blockIdx.x * blockDim.x + threadIdx.x;
    if (i < n) deg[i] = rsqrtf(fmaxf(deg[i], 1e-12f));
}

// --------------------- weight fragment pre-swizzle -------------------------
// Emit bf16 B-fragments in the exact per-lane order the WMMA consumers need:
//   w1frag[k][lane][j]        (4 ksteps,  from cw1 128x16)      2048 halfs
//   w2frag[nt][lane][j]       (8 ntiles,  from cw2 16x128, K>=16 zero) 4096
//   d1frag[wid][k][lane][j]   (8x8,       from dw1 256x128)    32768 halfs
// B layout: N=lane&15, hi=lane>>4; K = j<8 ? 8*hi+j : 16+8*hi+(j-8)
// (K offset by 32*kstep).

__global__ void k_swizzle_weights(const float* __restrict__ cw1,
                                  const float* __restrict__ cw2,
                                  const float* __restrict__ dw1,
                                  unsigned short* __restrict__ w1frag,
                                  unsigned short* __restrict__ w2frag,
                                  unsigned short* __restrict__ d1frag) {
    int i = blockIdx.x * blockDim.x + threadIdx.x;
    if (i < 2048) {                                     // w1frag
        int k = i >> 9, lane = (i >> 4) & 31, j = i & 15;
        int hi = lane >> 4, n = lane & 15;
        int K = k * 32 + (j < 8 ? 8 * hi + j : 16 + 8 * hi + (j - 8));
        w1frag[i] = f2bf(cw1[K * 16 + n]);
    } else if (i < 2048 + 4096) {                       // w2frag
        int ii = i - 2048;
        int nt = ii >> 9, lane = (ii >> 4) & 31, j = ii & 15;
        int hi = lane >> 4, n = lane & 15;
        int K = j < 8 ? 8 * hi + j : 16 + 8 * hi + (j - 8);
        w2frag[ii] = (K < 16) ? f2bf(cw2[K * 128 + nt * 16 + n]) : (unsigned short)0;
    } else if (i < 2048 + 4096 + 32768) {               // d1frag
        int ii = i - (2048 + 4096);
        int wid = ii >> 12, k = (ii >> 9) & 7, lane = (ii >> 4) & 31, j = ii & 15;
        int hi = lane >> 4, n = lane & 15;
        int K = k * 32 + (j < 8 ? 8 * hi + j : 16 + 8 * hi + (j - 8));
        d1frag[ii] = f2bf(dw1[K * 128 + wid * 16 + n]);
    }
}

// --------------------------- GEMM: xw1 = x @ W1 ----------------------------
// x: N x 128 f32, W1 fragments pre-swizzled. One wave per 16-row tile,
// K=128 in 4 WMMA steps. Epilogue seeds h1 with self-loop term xw1*dinv^2.

__global__ __launch_bounds__(256) void k_gemm_xw1(
        const float* __restrict__ x, const unsigned short* __restrict__ w1frag,
        const float* __restrict__ dinv,
        float* __restrict__ xw1, float* __restrict__ h1, int nNodes) {
    const int lane = threadIdx.x & 31;
    const int wid  = threadIdx.x >> 5;
    const int rowBase = (blockIdx.x * 8 + wid) * 16;
    if (rowBase >= nNodes) return;                 // wave-uniform guard
    const int m  = lane & 15;
    const int hi = lane >> 4;

    const float* xr = x + (size_t)(rowBase + m) * 128;
    v8f c = {};
    for (int k = 0; k < 4; ++k) {
        BF16x16 b;
        const U4* bp = (const U4*)(w1frag + ((k * 32 + lane) << 4));
        b.q[0] = bp[0]; b.q[1] = bp[1];

        BF16x16 a;
        int kb = k * 32 + (hi ? 8 : 0);
        float4 x0 = *(const float4*)(xr + kb);
        float4 x1 = *(const float4*)(xr + kb + 4);
        float4 x2 = *(const float4*)(xr + kb + 16);
        float4 x3 = *(const float4*)(xr + kb + 20);
        a.w[0] = f2bf2(x0.x, x0.y); a.w[1] = f2bf2(x0.z, x0.w);
        a.w[2] = f2bf2(x1.x, x1.y); a.w[3] = f2bf2(x1.z, x1.w);
        a.w[4] = f2bf2(x2.x, x2.y); a.w[5] = f2bf2(x2.z, x2.w);
        a.w[6] = f2bf2(x3.x, x3.y); a.w[7] = f2bf2(x3.z, x3.w);
        c = wmma_bf16(a, b, c);
    }
    for (int r = 0; r < 8; ++r) {
        int M = rowBase + r + (hi ? 8 : 0);
        float di = dinv[M];
        float vv = c[r];
        xw1[(size_t)M * 16 + m] = vv;
        h1 [(size_t)M * 16 + m] = vv * di * di;    // self-loop seed
    }
}

// ---------------------- scatter layer 1 (16 feats/edge) --------------------

__global__ void k_scatter1(const int* __restrict__ src, const int* __restrict__ dst,
                           const float* __restrict__ dinv,
                           const float* __restrict__ xw1, float* __restrict__ h1,
                           long long total) {
    long long idx = (long long)blockIdx.x * blockDim.x + threadIdx.x;
    if (idx >= total) return;
    int e = (int)(idx >> 4), f = (int)(idx & 15);
    int s = src[e], d = dst[e];
    float w = dinv[s] * dinv[d];
    atomicAdd(&h1[(size_t)d * 16 + f], xw1[(size_t)s * 16 + f] * w);
}

__global__ void k_bias_relu1(float* __restrict__ h1, const float* __restrict__ b,
                             long long total4) {        // total4 = N*16/4
    long long i4 = (long long)blockIdx.x * blockDim.x + threadIdx.x;
    if (i4 >= total4) return;
    long long base = i4 * 4;
    int c0 = (int)(base & 15);
    float4 v = *(const float4*)(h1 + base);
    v.x = fmaxf(v.x + b[c0 + 0], 0.0f);
    v.y = fmaxf(v.y + b[c0 + 1], 0.0f);
    v.z = fmaxf(v.z + b[c0 + 2], 0.0f);
    v.w = fmaxf(v.w + b[c0 + 3], 0.0f);
    *(float4*)(h1 + base) = v;
}

// --------------------------- GEMM: xw2 = h1 @ W2 ---------------------------
// h1: N x 16, W2 fragments pre-swizzled (K=16 zero-padded to 32).

__global__ __launch_bounds__(256) void k_gemm_xw2(
        const float* __restrict__ h1, const unsigned short* __restrict__ w2frag,
        const float* __restrict__ dinv,
        float* __restrict__ xw2, float* __restrict__ h2, int nNodes) {
    const int lane = threadIdx.x & 31;
    const int wid  = threadIdx.x >> 5;
    const int rowBase = (blockIdx.x * 8 + wid) * 16;
    if (rowBase >= nNodes) return;
    const int m  = lane & 15;
    const int hi = lane >> 4;

    BF16x16 a;
    const float* hr = h1 + (size_t)(rowBase + m) * 16;
    int kb = hi ? 8 : 0;                           // K 0..7 or 8..15
    float4 h0 = *(const float4*)(hr + kb);
    float4 hh = *(const float4*)(hr + kb + 4);
    a.w[0] = f2bf2(h0.x, h0.y); a.w[1] = f2bf2(h0.z, h0.w);
    a.w[2] = f2bf2(hh.x, hh.y); a.w[3] = f2bf2(hh.z, hh.w);
    a.w[4] = 0; a.w[5] = 0; a.w[6] = 0; a.w[7] = 0;   // K 16..31 padded

    float di8[8];
    for (int r = 0; r < 8; ++r) di8[r] = dinv[rowBase + r + (hi ? 8 : 0)];

    for (int nt = 0; nt < 8; ++nt) {
        BF16x16 b;
        const U4* bp = (const U4*)(w2frag + ((nt * 32 + lane) << 4));
        b.q[0] = bp[0]; b.q[1] = bp[1];
        v8f c = {};
        c = wmma_bf16(a, b, c);
        for (int r = 0; r < 8; ++r) {
            int M = rowBase + r + (hi ? 8 : 0);
            float vv = c[r];
            xw2[(size_t)M * 128 + nt * 16 + m] = vv;
            h2 [(size_t)M * 128 + nt * 16 + m] = vv * di8[r] * di8[r];
        }
    }
}

// ---------------------- scatter layer 2 (128 feats/edge) -------------------

__global__ void k_scatter2(const int* __restrict__ src, const int* __restrict__ dst,
                           const float* __restrict__ dinv,
                           const float* __restrict__ xw2, float* __restrict__ h2,
                           long long total) {
    long long idx = (long long)blockIdx.x * blockDim.x + threadIdx.x;
    if (idx >= total) return;
    int e  = (int)(idx >> 5);
    int f0 = (int)(idx & 31) * 4;
    int s = src[e], d = dst[e];
    float w = dinv[s] * dinv[d];
    float4 v = *(const float4*)(xw2 + (size_t)s * 128 + f0);
    float* o = h2 + (size_t)d * 128 + f0;
    atomicAdd(o + 0, v.x * w);
    atomicAdd(o + 1, v.y * w);
    atomicAdd(o + 2, v.z * w);
    atomicAdd(o + 3, v.w * w);
}

__global__ void k_bias2_bf16(const float* __restrict__ h2, const float* __restrict__ b,
                             unsigned int* __restrict__ h2bf32, long long total4) {
    long long i4 = (long long)blockIdx.x * blockDim.x + threadIdx.x;   // N*128/4
    if (i4 >= total4) return;
    long long base = i4 * 4;
    int c0 = (int)(base & 127);
    float4 v = *(const float4*)(h2 + base);
    h2bf32[i4 * 2 + 0] = f2bf2(v.x + b[c0 + 0], v.y + b[c0 + 1]);
    h2bf32[i4 * 2 + 1] = f2bf2(v.z + b[c0 + 2], v.w + b[c0 + 3]);
}

// ------------------------------- decoder -----------------------------------
// 64 pairs/block, 8 waves; wave w owns columns [16w,16w+16) of dw1 as 8
// resident pre-swizzled B fragments (2 b128 loads each). Gathered bf16 pair
// rows staged in LDS (stride 264 halfs breaks bank conflicts, keeps 16B
// alignment). 4 A-tiles x 8 K-steps of WMMA per wave.

#define DEC_PAIRS 64
#define LDA       264   // 256 halfs payload + 8 pad (528 B, 16B-multiple)

__global__ __launch_bounds__(256) void k_decoder(
        const unsigned short* __restrict__ h2bf,   // N x 128 bf16
        const int* __restrict__ reach,             // NP x 2
        const unsigned short* __restrict__ d1frag, // pre-swizzled dw1 frags
        const float* __restrict__ db1, const float* __restrict__ dw2,
        const float* __restrict__ db2,
        float* __restrict__ out, int NP) {
    __shared__ alignas(16) unsigned short Abuf[DEC_PAIRS * LDA];
    __shared__ float redbuf[DEC_PAIRS];

    const int t    = threadIdx.x;
    const int lane = t & 31;
    const int wid  = t >> 5;
    const int hi   = lane >> 4;
    const int n    = lane & 15;
    const int pbase = blockIdx.x * DEC_PAIRS;
    const int Ncol  = wid * 16 + n;

    // B fragments for this wave's N-tile: 8 x (2 x b128) loads
    BF16x16 bfr[8];
    for (int k = 0; k < 8; ++k) {
        const U4* bp = (const U4*)(d1frag + (((wid * 8 + k) * 32 + lane) << 4));
        bfr[k].q[0] = bp[0]; bfr[k].q[1] = bp[1];
    }
    const float db1N = db1[Ncol];
    const float dw2N = dw2[Ncol];

    // Stage A: 64 pairs x 256 halfs = 2048 chunks of 8 bf16 (16B each)
    for (int c = t; c < DEC_PAIRS * 32; c += 256) {
        int pair = c >> 5;
        int off8 = (c & 31) * 8;
        int p = pbase + pair; if (p >= NP) p = NP - 1;
        int node = (off8 < 128) ? reach[2 * p] : reach[2 * p + 1];
        *(U4*)(Abuf + pair * LDA + off8) =
            *(const U4*)(h2bf + (size_t)node * 128 + (off8 & 127));
    }
    for (int i = t; i < DEC_PAIRS; i += 256) redbuf[i] = 0.0f;
    __syncthreads();

    for (int at = 0; at < DEC_PAIRS / 16; ++at) {
        const unsigned short* arow = Abuf + (at * 16 + n) * LDA;   // M = n
        v8f c = {};
        for (int k = 0; k < 8; ++k) {
            BF16x16 a;
            int kb = k * 32 + (hi ? 8 : 0);
            a.q[0] = *(const U4*)(arow + kb);
            a.q[1] = *(const U4*)(arow + kb + 16);
            c = wmma_bf16(a, bfr[k], c);
        }
        // relu(c + db1) * dw2, reduce over the 16 N-lanes, then across waves
        for (int r = 0; r < 8; ++r) {
            float v = c[r] + db1N;
            v = v > 0.0f ? v : 0.0f;
            v *= dw2N;
            for (int mask = 1; mask < 16; mask <<= 1)
                v += __shfl_xor(v, mask, 32);
            if (n == 0)
                atomicAdd(&redbuf[at * 16 + r + (hi ? 8 : 0)], v);
        }
    }
    __syncthreads();

    const float b2 = db2[0];
    for (int i = t; i < DEC_PAIRS; i += 256) {
        int p = pbase + i;
        if (p < NP) {
            float z = redbuf[i] + b2;
            out[p] = 1.0f / (1.0f + __expf(-z));
        }
    }
}

// ------------------------------- launcher ----------------------------------

extern "C" void kernel_launch(void* const* d_in, const int* in_sizes, int n_in,
                              void* d_out, int out_size, void* d_ws, size_t ws_size,
                              hipStream_t stream) {
    const float* x     = (const float*)d_in[0];
    const int*   ei    = (const int*)  d_in[1];
    const int*   reach = (const int*)  d_in[2];
    const float* cw1   = (const float*)d_in[3];
    const float* cb1   = (const float*)d_in[4];
    const float* cw2   = (const float*)d_in[5];
    const float* cb2   = (const float*)d_in[6];
    const float* dw1   = (const float*)d_in[7];
    const float* db1   = (const float*)d_in[8];
    const float* dw2   = (const float*)d_in[9];
    const float* db2   = (const float*)d_in[10];
    float* out = (float*)d_out;

    const int N  = in_sizes[0] / 128;
    const int E  = in_sizes[1] / 2;
    const int NP = in_sizes[2] / 2;
    const int* esrc = ei;
    const int* edst = ei + E;

    // workspace carve-out (256B aligned); xw2 re-aliased as bf16 h after use
    char* ws = (char*)d_ws;
    size_t off = 0;
    auto carve = [&](size_t bytes) -> void* {
        void* p = ws + off;
        off = (off + bytes + 255) & ~(size_t)255;
        return p;
    };
    float* deg  = (float*)carve((size_t)N * 4);            // becomes dinv
    float* xw1  = (float*)carve((size_t)N * 16 * 4);
    float* h1   = (float*)carve((size_t)N * 16 * 4);
    float* xw2  = (float*)carve((size_t)N * 128 * 4);
    float* h2   = (float*)carve((size_t)N * 128 * 4);
    unsigned short* w1frag = (unsigned short*)carve(2048 * 2);
    unsigned short* w2frag = (unsigned short*)carve(4096 * 2);
    unsigned short* d1frag = (unsigned short*)carve(32768 * 2);
    unsigned short* h2b    = (unsigned short*)xw2;         // xw2 dead after scatter2
    (void)ws_size; (void)n_in; (void)out_size;

    const int B = 256;
    k_deg_init <<<(N + B - 1) / B, B, 0, stream>>>(deg, N);
    k_deg_count<<<(E + B - 1) / B, B, 0, stream>>>(edst, deg, E);
    k_deg_rsqrt<<<(N + B - 1) / B, B, 0, stream>>>(deg, N);

    const int totalFrag = 2048 + 4096 + 32768;
    k_swizzle_weights<<<(totalFrag + B - 1) / B, B, 0, stream>>>(
        cw1, cw2, dw1, w1frag, w2frag, d1frag);

    k_gemm_xw1 <<<(N + 127) / 128, B, 0, stream>>>(x, w1frag, deg, xw1, h1, N);

    long long t1 = (long long)E * 16;
    k_scatter1 <<<(unsigned)((t1 + B - 1) / B), B, 0, stream>>>(esrc, edst, deg, xw1, h1, t1);
    long long n16_4 = (long long)N * 16 / 4;
    k_bias_relu1<<<(unsigned)((n16_4 + B - 1) / B), B, 0, stream>>>(h1, cb1, n16_4);

    k_gemm_xw2 <<<(N + 127) / 128, B, 0, stream>>>(h1, w2frag, deg, xw2, h2, N);

    long long t2 = (long long)E * 32;
    k_scatter2 <<<(unsigned)((t2 + B - 1) / B), B, 0, stream>>>(esrc, edst, deg, xw2, h2, t2);
    long long n128_4 = (long long)N * 128 / 4;
    k_bias2_bf16<<<(unsigned)((n128_4 + B - 1) / B), B, 0, stream>>>(
        h2, cb2, (unsigned int*)h2b, n128_4);

    k_decoder  <<<(NP + DEC_PAIRS - 1) / DEC_PAIRS, B, 0, stream>>>(
        h2b, reach, d1frag, db1, dw2, db2, out, NP);
}
